// MultiSourceCopyGenerator_17102559772958
// MI455X (gfx1250) — matrix operand order repaired and checked by reading, not compiled, not run on hardware
//
#include <hip/hip_runtime.h>
#include <hip/hip_bf16.h>

// ---------------------------------------------------------------------------
// Problem constants: B=32, T=64, S=400, D=1024, V=32000, E=600
// ---------------------------------------------------------------------------
#define BT     2048
#define DDIM   1024
#define VOCAB  32000
#define SLEN   400
#define BBATCH 32
#define EVOC   600
#define OUTC   (VOCAB + EVOC)
#define PAD_IDX 1

// GEMM tiling
#define BLK_M  128
#define BLK_N  128
#define KSTEP  64
#define LDSROW 72          // 64 K-elems + 8 pad (144B row stride, conflict-free)

typedef __attribute__((ext_vector_type(16))) __bf16 bf16x16;
typedef __attribute__((ext_vector_type(8)))  float  v8f;
typedef int v4i __attribute__((vector_size(16)));   // matches async-LDS builtin proto

union FragBits { uint4 u[2]; bf16x16 v; };

#define AS1 __attribute__((address_space(1)))
#define AS3 __attribute__((address_space(3)))

#if defined(__has_builtin)
#if __has_builtin(__builtin_amdgcn_global_load_async_to_lds_b128) && \
    __has_builtin(__builtin_amdgcn_s_wait_asynccnt)
#define GEMM_ASYNC 1
#endif
#endif
#ifndef GEMM_ASYNC
#define GEMM_ASYNC 0
#endif

// ---------------------------------------------------------------------------
// f32 -> bf16 (round-to-nearest-even)
// ---------------------------------------------------------------------------
__device__ __forceinline__ unsigned short f2bf(float f) {
  unsigned int u = __float_as_uint(f);
  u = (u + 0x7FFFu + ((u >> 16) & 1u)) >> 16;
  return (unsigned short)u;
}

__global__ __launch_bounds__(256) void cvt_bf16_kernel(
    const float* __restrict__ in, unsigned short* __restrict__ out, int n) {
  int i = (blockIdx.x * 256 + threadIdx.x) * 4;
  if (i >= n) return;
  float4 v = *(const float4*)(in + i);
  ushort4 r;
  r.x = f2bf(v.x); r.y = f2bf(v.y); r.z = f2bf(v.z); r.w = f2bf(v.w);
  *(ushort4*)(out + i) = r;
}

// ---------------------------------------------------------------------------
// p_copy = sigmoid(hidden . Wc + bc), one wave32 per row
// ---------------------------------------------------------------------------
__global__ __launch_bounds__(256) void pcopy_kernel(
    const float* __restrict__ hidden, const float* __restrict__ Wc,
    const float* __restrict__ bc, float* __restrict__ pcopy) {
  const int wave = threadIdx.x >> 5, lane = threadIdx.x & 31;
  const int row = blockIdx.x * 8 + wave;
  const float* h = hidden + (size_t)row * DDIM;
  float s = 0.f;
  for (int k = lane * 4; k < DDIM; k += 128) {
    float4 hv = *(const float4*)(h + k);
    float4 wv = *(const float4*)(Wc + k);
    s += hv.x * wv.x + hv.y * wv.y + hv.z * wv.z + hv.w * wv.w;
  }
  #pragma unroll
  for (int o = 16; o > 0; o >>= 1) s += __shfl_down(s, o, 32);
  if (lane == 0) pcopy[row] = 1.f / (1.f + __expf(-(s + bc[0])));
}

// ---------------------------------------------------------------------------
// Fragment helpers (CDNA5 ISA 7.12.2 layouts, wave32)
//  A 16x32 bf16: lane<16 -> row=lane,    K = {0..7, 16..23}
//                lane>=16 -> row=lane-16, K = {8..15, 24..31}
//  B 32x16 bf16: lane<16 -> col=lane,    K = {0..7, 16..23} ; lane>=16 -> K = {8..15, 24..31}
// ---------------------------------------------------------------------------
__device__ __forceinline__ bf16x16 load_fragA(const unsigned short* row, int kb, int lhi) {
  const int klo = lhi ? 8 : 0;
  FragBits f;
  f.u[0] = *(const uint4*)(row + kb + klo);
  f.u[1] = *(const uint4*)(row + kb + klo + 16);
  return f.v;
}

__device__ __forceinline__ bf16x16 lds_fragB(const unsigned short* ldsbuf, int n_local,
                                             int kk, int lhi) {
  const unsigned short* p = ldsbuf + n_local * LDSROW + kk + (lhi ? 8 : 0);
  FragBits f;
  f.u[0] = *(const uint4*)p;
  f.u[1] = *(const uint4*)(p + 16);
  return f.v;
}

// Stage a 128xKSTEP bf16 W tile into LDS (16KB payload -> 64B per thread)
__device__ __forceinline__ void stage_B_tile(
    const unsigned short* __restrict__ Wbf, unsigned short* ldsbuf,
    int nbase, int kb, int tid) {
  const int r = tid >> 1;             // row 0..127 within tile
  const int h = (tid & 1) * 32;       // element offset 0 or 32
  const unsigned short* g = Wbf + (size_t)(nbase + r) * DDIM + kb + h;
  unsigned short* l = ldsbuf + r * LDSROW + h;
#if GEMM_ASYNC
  #pragma unroll
  for (int c = 0; c < 4; ++c) {
    __builtin_amdgcn_global_load_async_to_lds_b128(
        (AS1 v4i*)(g + c * 8), (AS3 v4i*)(l + c * 8), 0, 0);
  }
#else
  #pragma unroll
  for (int c = 0; c < 4; ++c) {
    uint4 v = *(const uint4*)(g + c * 8);
    *(uint4*)(l + c * 8) = v;
  }
#endif
}

__device__ __forceinline__ void wait_stage() {
#if GEMM_ASYNC
  __builtin_amdgcn_s_wait_asynccnt(0);
#endif
}

#define MMA(acc, a, b) \
  acc = __builtin_amdgcn_wmma_f32_16x16x32_bf16(false, (a), false, (b), (short)0, (acc), false, false)

// Epilogue store of one 16x16 C tile (by value; no register-array indirection)
__device__ __forceinline__ void store_tile(v8f acc, int row_lo, int ncol,
                                           const float* __restrict__ bias,
                                           float* __restrict__ out) {
  const float bv = bias[ncol];
  #pragma unroll
  for (int v = 0; v < 8; ++v) {
    float val = acc[v] + bv;
    if (ncol == PAD_IDX) val = -__builtin_inff();
    out[(size_t)(row_lo + v) * OUTC + ncol] = val;
  }
}

// ---------------------------------------------------------------------------
// bf16 WMMA GEMM: logits = Hbf @ Wbf^T + bias, PAD col -> -inf.
// Block 256 thr = 8 waves (4 wm x 2 wn). Wave tile 32M x 64N (8 named v8f acc).
// Block tile 128M x 128N; W tile double-buffered through LDS (async-to-LDS).
// Grid: (2048/128, 32000/128) = (16, 250).
// ---------------------------------------------------------------------------
__global__ __launch_bounds__(256) void gemm_bf16_wmma_kernel(
    const unsigned short* __restrict__ Hbf, const unsigned short* __restrict__ Wbf,
    const float* __restrict__ bias, float* __restrict__ out) {
  __shared__ __align__(16) unsigned short ldsB[2][BLK_N * LDSROW];

  const int tid  = threadIdx.x;
  const int wave = tid >> 5, lane = tid & 31;
  const int wm = wave >> 1, wn = wave & 1;
  const int lr  = lane & 15;
  const int lhi = lane & 16;
  const int mbase = blockIdx.x * BLK_M;
  const int nbase = blockIdx.y * BLK_N;
  const int nloc  = wn * 64 + lr;

  const unsigned short* arow0 = Hbf + (size_t)(mbase + wm * 32 + lr) * DDIM;
  const unsigned short* arow1 = arow0 + (size_t)16 * DDIM;

  v8f c00 = {}, c01 = {}, c02 = {}, c03 = {};
  v8f c10 = {}, c11 = {}, c12 = {}, c13 = {};

  stage_B_tile(Wbf, ldsB[0], nbase, 0, tid);
  int buf = 0;

  #pragma unroll 1
  for (int s = 0; s < DDIM / KSTEP; ++s) {
    wait_stage();
    __syncthreads();
    if (s + 1 < DDIM / KSTEP)
      stage_B_tile(Wbf, ldsB[buf ^ 1], nbase, (s + 1) * KSTEP, tid);

    const unsigned short* lcur = ldsB[buf];
    const int kb = s * KSTEP;
    #pragma unroll
    for (int half = 0; half < 2; ++half) {
      const int kk = half * 32;
      bf16x16 a0 = load_fragA(arow0, kb + kk, lhi);
      bf16x16 a1 = load_fragA(arow1, kb + kk, lhi);
      bf16x16 b;
      b = lds_fragB(lcur, nloc +  0, kk, lhi); MMA(c00, a0, b); MMA(c10, a1, b);
      b = lds_fragB(lcur, nloc + 16, kk, lhi); MMA(c01, a0, b); MMA(c11, a1, b);
      b = lds_fragB(lcur, nloc + 32, kk, lhi); MMA(c02, a0, b); MMA(c12, a1, b);
      b = lds_fragB(lcur, nloc + 48, kk, lhi); MMA(c03, a0, b); MMA(c13, a1, b);
    }
    __syncthreads();
    buf ^= 1;
  }

  // Epilogue: C/D layout -> VGPR j: lane<16 M=j, lane>=16 M=j+8; N = lane&15
  const int row0 = mbase + wm * 32 + (lhi ? 8 : 0);       // sub-tile 0 rows
  const int row1 = row0 + 16;                              // sub-tile 1 rows
  const int ncol0 = nbase + wn * 64 + lr;
  store_tile(c00, row0, ncol0 +  0, bias, out);
  store_tile(c01, row0, ncol0 + 16, bias, out);
  store_tile(c02, row0, ncol0 + 32, bias, out);
  store_tile(c03, row0, ncol0 + 48, bias, out);
  store_tile(c10, row1, ncol0 +  0, bias, out);
  store_tile(c11, row1, ncol0 + 16, bias, out);
  store_tile(c12, row1, ncol0 + 32, bias, out);
  store_tile(c13, row1, ncol0 + 48, bias, out);
}

// ---------------------------------------------------------------------------
// Online softmax row stats (single read): rowmax, rowsumexp. 1 block/row.
// ---------------------------------------------------------------------------
__global__ __launch_bounds__(256) void rowstat_kernel(
    const float* __restrict__ out, float* __restrict__ rowmax, float* __restrict__ rowsum) {
  const int row = blockIdx.x, tid = threadIdx.x;
  const float* p = out + (size_t)row * OUTC;
  float m = -__builtin_inff(), s = 0.f;
  for (int i = tid * 4; i < VOCAB; i += 1024) {
    float4 v = *(const float4*)(p + i);
    float lm = fmaxf(fmaxf(v.x, v.y), fmaxf(v.z, v.w));
    if (lm > m) { s *= __expf(m - lm); m = lm; }
    s += __expf(v.x - m) + __expf(v.y - m) + __expf(v.z - m) + __expf(v.w - m);
  }
  __shared__ float sm[256], ss[256];
  sm[tid] = m; ss[tid] = s;
  __syncthreads();
  for (int w = 128; w > 0; w >>= 1) {
    if (tid < w) {
      float m2 = sm[tid + w], s2 = ss[tid + w];
      float nm = fmaxf(sm[tid], m2);
      ss[tid] = ss[tid] * __expf(sm[tid] - nm) + s2 * __expf(m2 - nm);
      sm[tid] = nm;
    }
    __syncthreads();
  }
  if (tid == 0) { rowmax[row] = sm[0]; rowsum[row] = ss[0]; }
}

// ---------------------------------------------------------------------------
// In-place finalize: out = exp(x - max)/sum * (1 - p_copy)
// ---------------------------------------------------------------------------
__global__ __launch_bounds__(256) void finalize_kernel(
    float* __restrict__ out, const float* __restrict__ rowmax,
    const float* __restrict__ rowsum, const float* __restrict__ pcopy) {
  const int row = blockIdx.y;
  const int i = (blockIdx.x * 256 + threadIdx.x) * 4;
  if (i >= VOCAB) return;
  const float m = rowmax[row];
  const float scale = (1.f - pcopy[row]) / rowsum[row];
  float* p = out + (size_t)row * OUTC + i;
  float4 v = *(const float4*)p;
  v.x = __expf(v.x - m) * scale;
  v.y = __expf(v.y - m) * scale;
  v.z = __expf(v.z - m) * scale;
  v.w = __expf(v.w - m) * scale;
  *(float4*)p = v;
}

// ---------------------------------------------------------------------------
// idx[s*32+b] = argmax_e src_map[s,b,e] (first max on ties)
// ---------------------------------------------------------------------------
__global__ __launch_bounds__(256) void argmax_kernel(
    const float* __restrict__ src_map, int* __restrict__ idx) {
  const int wid = blockIdx.x * 8 + (threadIdx.x >> 5);
  const int lane = threadIdx.x & 31;
  const float* p = src_map + (size_t)wid * EVOC;
  float best = -__builtin_inff(); int bi = 0x7fffffff;
  for (int e = lane; e < EVOC; e += 32) {
    float v = p[e];
    if (v > best) { best = v; bi = e; }
  }
  #pragma unroll
  for (int o = 16; o > 0; o >>= 1) {
    float ob = __shfl_down(best, o, 32);
    int   oi = __shfl_down(bi,   o, 32);
    if (ob > best || (ob == best && oi < bi)) { best = ob; bi = oi; }
  }
  if (lane == 0) idx[wid] = bi;
}

// ---------------------------------------------------------------------------
// copy_prob scatter: out[r, 32000+c] += attn[r,s]*p_copy[r] where idx[s,r%32]==c
// ---------------------------------------------------------------------------
__global__ __launch_bounds__(128) void copy_scatter_kernel(
    const float* __restrict__ attn, const float* __restrict__ pcopy,
    const int* __restrict__ idx, float* __restrict__ out) {
  const int row = blockIdx.x;
  const int bcol = row & (BBATCH - 1);
  __shared__ float bins[EVOC];
  for (int c = threadIdx.x; c < EVOC; c += 128) bins[c] = 0.f;
  __syncthreads();
  const float pc = pcopy[row];
  const float* a = attn + (size_t)row * SLEN;
  for (int s = threadIdx.x; s < SLEN; s += 128) {
    atomicAdd(&bins[idx[s * BBATCH + bcol]], a[s] * pc);
  }
  __syncthreads();
  float* o = out + (size_t)row * OUTC + VOCAB;
  for (int c = threadIdx.x; c < EVOC; c += 128) o[c] = bins[c];
}

// ---------------------------------------------------------------------------
// Launch
// ---------------------------------------------------------------------------
extern "C" void kernel_launch(void* const* d_in, const int* in_sizes, int n_in,
                              void* d_out, int out_size, void* d_ws, size_t ws_size,
                              hipStream_t stream) {
  const float* hidden  = (const float*)d_in[0];   // (2048, 1024)
  const float* attn    = (const float*)d_in[1];   // (2048, 400)
  const float* src_map = (const float*)d_in[2];   // (400, 32, 600)
  const float* W       = (const float*)d_in[3];   // (32000, 1024)
  const float* bias    = (const float*)d_in[4];   // (32000,)
  const float* Wc      = (const float*)d_in[5];   // (1, 1024)
  const float* bc      = (const float*)d_in[6];   // (1,)
  float* out = (float*)d_out;                     // (2048, 32600)

  char* ws = (char*)d_ws;
  unsigned short* Wbf = (unsigned short*)ws;
  unsigned short* Hbf = (unsigned short*)(ws + (size_t)VOCAB * DDIM * 2);
  float* rowmax = (float*)(ws + (size_t)VOCAB * DDIM * 2 + (size_t)BT * DDIM * 2);
  float* rowsum = rowmax + BT;
  float* pcopy  = rowsum + BT;
  int*   idx    = (int*)(pcopy + BT);

  cvt_bf16_kernel<<<(VOCAB * DDIM) / 1024, 256, 0, stream>>>(W, Wbf, VOCAB * DDIM);
  cvt_bf16_kernel<<<(BT * DDIM) / 1024, 256, 0, stream>>>(hidden, Hbf, BT * DDIM);

  pcopy_kernel<<<BT / 8, 256, 0, stream>>>(hidden, Wc, bc, pcopy);

  gemm_bf16_wmma_kernel<<<dim3(BT / BLK_M, VOCAB / BLK_N), 256, 0, stream>>>(Hbf, Wbf, bias, out);

  rowstat_kernel<<<BT, 256, 0, stream>>>(out, rowmax, rowsum);

  finalize_kernel<<<dim3((VOCAB / 4 + 255) / 256, BT), 256, 0, stream>>>(out, rowmax, rowsum, pcopy);

  argmax_kernel<<<(SLEN * BBATCH) / 8, 256, 0, stream>>>(src_map, idx);

  copy_scatter_kernel<<<BT, 128, 0, stream>>>(attn, pcopy, idx, out);
}
